// TokenFormerBlock_64029372449450
// MI455X (gfx1250) — matrix operand order repaired and checked by reading, not compiled, not run on hardware
//
#include <hip/hip_runtime.h>
#include <hip/hip_bf16.h>
#include <math.h>
#include <stdint.h>

// ---------------------------------------------------------------------------
// CDNA5 (gfx1250) TokenFormer block: bf16 WMMA everywhere, f32 accumulate.
// All GEMMs use the NT form (K contiguous on both operands); val weights are
// transposed once per launch. LDS staging is double-buffered and uses
// GLOBAL_LOAD_ASYNC_TO_LDS_B128 (ASYNCcnt) when the toolchain exposes it.
// ---------------------------------------------------------------------------

typedef __attribute__((ext_vector_type(16))) __bf16 v16bf;
typedef __attribute__((ext_vector_type(8)))  float  v8f;

// The async-load builtin's pointer params are gcc-style <4 x i32> vectors in
// AS1 (printed __device__) / AS3 (printed __shared__). Match them exactly.
typedef int v4i_gcc __attribute__((vector_size(16)));
typedef __attribute__((address_space(1))) v4i_gcc as1_v4i;
typedef __attribute__((address_space(3))) v4i_gcc as3_v4i;

struct alignas(16) U128 { unsigned int w[4]; };

union Frag16 { v16bf v; U128 u[2]; };

#define DEVINL __device__ __forceinline__

#if __has_builtin(__builtin_amdgcn_global_load_async_to_lds_b128)
#define HAVE_ASYNC_LDS 1
#else
#define HAVE_ASYNC_LDS 0
#endif

// 16-byte global -> LDS copy. Async (no VGPR bounce, tracked by ASYNCcnt)
// when available; plain vector copy otherwise.
DEVINL void copy16_g2l(void* lds_dst, const void* g_src) {
#if HAVE_ASYNC_LDS
  __builtin_amdgcn_global_load_async_to_lds_b128(
      (as1_v4i*)(uintptr_t)g_src,
      (as3_v4i*)(unsigned int)(uintptr_t)lds_dst,
      0, 0);
#else
  *(U128*)lds_dst = *(const U128*)g_src;
#endif
}

DEVINL void async_join() {
#if HAVE_ASYNC_LDS
#if __has_builtin(__builtin_amdgcn_s_wait_asynccnt)
  __builtin_amdgcn_s_wait_asynccnt(0);
#else
  asm volatile("s_wait_asynccnt 0x0" ::: "memory");
#endif
#endif
}

DEVINL v8f zero8() { v8f z = {0.f,0.f,0.f,0.f,0.f,0.f,0.f,0.f}; return z; }

DEVINL v8f wmma_bf16(v16bf a, v16bf b, v8f c) {
  // D = A(16x32 bf16) * B(32x16 bf16) + C(16x16 f32)
  return __builtin_amdgcn_wmma_f32_16x16x32_bf16(
      /*neg_a=*/false, a, /*neg_b=*/false, b,
      /*c_mod=*/(short)0, c, /*reuse_a=*/false, /*reuse_b=*/false);
}

// A fragment (16x32, bf16) gathered from a row-major [m][k] tile (ld in elems).
// ISA layout: lane holds row m=lane&15; i<8 -> k=off0+i ; i>=8 -> k=16+off0+(i-8)
// with off0 = (lane&16)>>1.
DEVINL v16bf frag_a(const __bf16* tile, int ld, int lane) {
  const __bf16* p = tile + (size_t)(lane & 15) * ld + ((lane & 16) >> 1);
  Frag16 f;
  f.u[0] = *(const U128*)(p);
  f.u[1] = *(const U128*)(p + 16);
  return f.v;
}

// B fragment (32x16, bf16) gathered from a [n][k] tile (ld in elems).
// ISA layout: lane holds col n=lane&15; i<8 -> k=off0+i ; i>=8 -> k=16+off0+(i-8).
DEVINL v16bf frag_b(const __bf16* tile, int ld, int lane) {
  const __bf16* p = tile + (size_t)(lane & 15) * ld + ((lane & 16) >> 1);
  Frag16 f;
  f.u[0] = *(const U128*)(p);
  f.u[1] = *(const U128*)(p + 16);
  return f.v;
}

DEVINL float gelu_exact(float x) {
  return 0.5f * x * (1.0f + erff(x * 0.70710678118654752f));
}

DEVINL float block_reduce_sum(float v, float* red) {
  const int tid = threadIdx.x;
  red[tid] = v;
  __syncthreads();
  for (int s = 128; s > 0; s >>= 1) {
    if (tid < s) red[tid] += red[tid + s];
    __syncthreads();
  }
  float r = red[0];
  __syncthreads();
  return r;
}

// ---------------------------------------------------------------------------
// bf16 WMMA GEMM (NT):  C(MxN,f32) = alpha * A(MxK) * B^T, B is (N x K)
// row-major. Macro tile 64x128, BK=32, double-buffered LDS with async
// global->LDS staging. 256 threads = 8 waves, each wave owns 32x32 of C.
// ---------------------------------------------------------------------------
__global__ __launch_bounds__(256) void gemm_bf16_nt_kernel(
    const __bf16* __restrict__ A, const __bf16* __restrict__ Bm,
    float* __restrict__ C, int N, int K, float alpha) {
  __shared__ __align__(16) __bf16 Asm[2][64][32];
  __shared__ __align__(16) __bf16 Bsm[2][128][32];   // [n][k]
  const int tid  = threadIdx.x;
  const int lane = tid & 31;
  const int w    = tid >> 5;
  const int wm   = (w & 1) * 32;
  const int wn   = (w >> 1) * 32;
  const int m0   = blockIdx.y * 64;
  const int n0   = blockIdx.x * 128;

  v8f acc[2][2];
  acc[0][0] = zero8(); acc[0][1] = zero8();
  acc[1][0] = zero8(); acc[1][1] = zero8();

  const int arow  = tid >> 2;          // 0..63
  const int akoff = (tid & 3) * 8;     // 0..24

  // Stage one 32-wide K-slice of A (64x32) and B (128x32) into buffer `buf`.
  auto stage = [&](int buf, int kt) {
    copy16_g2l(&Asm[buf][arow][akoff],
               &A[(size_t)(m0 + arow) * K + kt + akoff]);
    copy16_g2l(&Bsm[buf][arow][akoff],
               &Bm[(size_t)(n0 + arow) * K + kt + akoff]);
    copy16_g2l(&Bsm[buf][arow + 64][akoff],
               &Bm[(size_t)(n0 + arow + 64) * K + kt + akoff]);
  };

  const int nk = K >> 5;
  stage(0, 0);
  async_join();
  __syncthreads();

  for (int kt = 0; kt < nk; ++kt) {
    const int cur = kt & 1;
    if (kt + 1 < nk) stage(cur ^ 1, (kt + 1) << 5);   // prefetch next slice

    v16bf a0 = frag_a(&Asm[cur][wm][0],      32, lane);
    v16bf a1 = frag_a(&Asm[cur][wm + 16][0], 32, lane);
    v16bf b0 = frag_b(&Bsm[cur][wn][0],      32, lane);
    v16bf b1 = frag_b(&Bsm[cur][wn + 16][0], 32, lane);
    acc[0][0] = wmma_bf16(a0, b0, acc[0][0]);
    acc[0][1] = wmma_bf16(a0, b1, acc[0][1]);
    acc[1][0] = wmma_bf16(a1, b0, acc[1][0]);
    acc[1][1] = wmma_bf16(a1, b1, acc[1][1]);

    async_join();      // next slice resident
    __syncthreads();   // ...and all waves done reading `cur`
  }

  // Epilogue: D layout m = r + ((lane&16)>>1), n = lane&15.
  const int cn  = n0 + wn + (lane & 15);
  const int cmb = m0 + wm + ((lane & 16) >> 1);
#pragma unroll
  for (int i = 0; i < 2; ++i)
#pragma unroll
    for (int r = 0; r < 8; ++r) {
      int m = cmb + i * 16 + r;
      C[(size_t)m * N + cn]      = acc[i][0][r] * alpha;
      C[(size_t)m * N + cn + 16] = acc[i][1][r] * alpha;
    }
}

// ---------------------------------------------------------------------------
// Row-wise gelu + L2 norm: out_bf16 = gelu(in) * sqrt(P) / ||gelu(in)||_2
// ---------------------------------------------------------------------------
__global__ __launch_bounds__(256) void gelu_l2norm_kernel(
    const float* __restrict__ in, __bf16* __restrict__ out, int P) {
  __shared__ float red[256];
  const int row = blockIdx.x;
  const int tid = threadIdx.x;
  const float* r = in + (size_t)row * P;
  const int nc = P >> 8;           // P/256, max 16
  float g[16];
  float ss = 0.f;
  for (int i = 0; i < nc; ++i) {
    float gg = gelu_exact(r[i * 256 + tid]);
    g[i] = gg;
    ss += gg * gg;
  }
  float tot = block_reduce_sum(ss, red);
  float scale = sqrtf((float)P) * rsqrtf(tot);
  for (int i = 0; i < nc; ++i)
    out[(size_t)row * P + i * 256 + tid] = (__bf16)(g[i] * scale);
}

// ---------------------------------------------------------------------------
// LayerNorm over H=1024 -> bf16
// ---------------------------------------------------------------------------
__global__ __launch_bounds__(256) void layernorm_bf16_kernel(
    const float* __restrict__ x, const float* __restrict__ wt,
    const float* __restrict__ bs, __bf16* __restrict__ out) {
  __shared__ float red[256];
  const int row = blockIdx.x;
  const int tid = threadIdx.x;
  const float* r = x + (size_t)row * 1024;
  float v[4];
  float s = 0.f, s2 = 0.f;
#pragma unroll
  for (int i = 0; i < 4; ++i) {
    float xv = r[i * 256 + tid];
    v[i] = xv; s += xv; s2 += xv * xv;
  }
  float S1 = block_reduce_sum(s,  red);
  float S2 = block_reduce_sum(s2, red);
  float mu  = S1 * (1.f / 1024.f);
  float var = S2 * (1.f / 1024.f) - mu * mu;
  float rs  = rsqrtf(var + 1e-5f);
#pragma unroll
  for (int i = 0; i < 4; ++i) {
    int c = i * 256 + tid;
    out[(size_t)row * 1024 + c] = (__bf16)((v[i] - mu) * rs * wt[c] + bs[c]);
  }
}

// ---------------------------------------------------------------------------
// RoPE (ROT=16 of HD=64) on q,k and repack q/k/v: (B,S,NH,HD)f32 ->
// (B,NH,S,HD)bf16.  B=2,S=2048,NH=16,HD=64.
// ---------------------------------------------------------------------------
__global__ __launch_bounds__(256) void rope_pack_kernel(
    const float* __restrict__ qf, const float* __restrict__ kf,
    const float* __restrict__ vf, __bf16* __restrict__ qh,
    __bf16* __restrict__ kh, __bf16* __restrict__ vh) {
  size_t idx = (size_t)blockIdx.x * 256 + threadIdx.x;
  if (idx >= (size_t)2 * 2048 * 16 * 64) return;
  int d = idx & 63;
  int h = (int)((idx >> 6) & 15);
  int s = (int)((idx >> 10) & 2047);
  int b = (int)(idx >> 21);
  size_t src = ((size_t)(b * 2048 + s)) * 1024 + h * 64 + d;
  size_t dst = (((size_t)(b * 16 + h)) * 2048 + s) * 64 + d;
  float qv = qf[src], kv = kf[src];
  if (d < 16) {
    int j = d & 7;
    float inv = powf(10000.f, -(float)(2 * j) / 16.f);
    float ang = (float)s * inv;
    float c = cosf(ang), sn = sinf(ang);
    if (d < 8) { qv = qv * c - qf[src + 8] * sn; kv = kv * c - kf[src + 8] * sn; }
    else       { qv = qv * c + qf[src - 8] * sn; kv = kv * c + kf[src - 8] * sn; }
  }
  qh[dst] = (__bf16)qv;
  kh[dst] = (__bf16)kv;
  vh[dst] = (__bf16)vf[src];
}

// ---------------------------------------------------------------------------
// Flash-style causal attention with gelu_l2_norm weights.
// Grid: (S/64, B*NH), 128 threads (4 waves), each wave owns a 16-query tile.
// Per 32-key chunk: 4 WMMAs for scores (2 key-tiles x K=64) + 4 WMMAs for o.
// gelu_l2_norm's scalar row scale is applied once at the end:
//   o = (sum_k g*v) * sqrt(S) / sqrt(sum_k g^2)
// Output ob in (B,S,H) bf16 so proj-pattention consumes it directly.
// ---------------------------------------------------------------------------
__global__ __launch_bounds__(128) void attention_kernel(
    const __bf16* __restrict__ qh, const __bf16* __restrict__ kh,
    const __bf16* __restrict__ vh, __bf16* __restrict__ ob) {
  __shared__ __align__(16) __bf16 g_lds[4][16 * 32];
  __shared__ __align__(16) __bf16 vt[4][64 * 32];   // [hd][key], transposed
  const int w    = threadIdx.x >> 5;
  const int lane = threadIdx.x & 31;
  const int bh   = blockIdx.y;
  const int qbase = (blockIdx.x * 4 + w) * 16;
  const size_t seqbase = (size_t)bh * 2048;

  // q fragments for both K-halves (HD=64 -> kc=0: k 0..31, kc=1: k 32..63)
  v16bf aq[2];
  {
    const __bf16* qrow =
        qh + (seqbase + qbase + (lane & 15)) * 64 + ((lane & 16) >> 1);
    Frag16 f0, f1;
    f0.u[0] = *(const U128*)(qrow);
    f0.u[1] = *(const U128*)(qrow + 16);
    f1.u[0] = *(const U128*)(qrow + 32);
    f1.u[1] = *(const U128*)(qrow + 48);
    aq[0] = f0.v; aq[1] = f1.v;
  }

  v8f oacc[4];
#pragma unroll
  for (int i = 0; i < 4; ++i) oacc[i] = zero8();
  float ssa[8] = {0.f,0.f,0.f,0.f,0.f,0.f,0.f,0.f};

  const int kbmax = (qbase + 15) >> 5;       // causal frontier (32-key chunks)
  for (int kb = 0; kb <= kbmax; ++kb) {
    // Stage v chunk (32 keys x 64 hd) transposed into LDS [hd][key].
    const __bf16* vrow = vh + (seqbase + kb * 32 + lane) * 64;
#pragma unroll 8
    for (int d = 0; d < 64; ++d) vt[w][d * 32 + lane] = vrow[d];
    __builtin_amdgcn_wave_barrier();

#pragma unroll
    for (int t = 0; t < 2; ++t) {
      v8f sc = zero8();
      const int key = kb * 32 + t * 16 + (lane & 15);
#pragma unroll
      for (int kc = 0; kc < 2; ++kc) {
        // B fragment straight from global k rows (K-contiguous, 32B/lane).
        const __bf16* kp =
            kh + (seqbase + key) * 64 + kc * 32 + ((lane & 16) >> 1);
        Frag16 fb;
        fb.u[0] = *(const U128*)kp;
        fb.u[1] = *(const U128*)(kp + 16);
        sc = wmma_bf16(aq[kc], fb.v, sc);
      }
      // g = gelu(score/8) with causal zeroing; stage in A-layout LDS tile.
#pragma unroll
      for (int r = 0; r < 8; ++r) {
        int mm = r + ((lane & 16) >> 1);
        float sv = sc[r] * 0.125f;                    // 1/sqrt(HD)
        float g  = (key <= qbase + mm) ? gelu_exact(sv) : 0.f;
        ssa[r] += g * g;
        g_lds[w][mm * 32 + t * 16 + (lane & 15)] = (__bf16)g;
      }
    }
    __builtin_amdgcn_wave_barrier();

    v16bf ag = frag_a(&g_lds[w][0], 32, lane);
#pragma unroll
    for (int nt = 0; nt < 4; ++nt) {
      v16bf bv = frag_b(&vt[w][nt * 16 * 32], 32, lane);
      oacc[nt] = wmma_bf16(ag, bv, oacc[nt]);
    }
  }

  // Reduce ss across the 16 lanes sharing each row-half (xor stays in-group).
#pragma unroll
  for (int r = 0; r < 8; ++r) {
    float s = ssa[r];
    s += __shfl_xor(s, 1, 32);
    s += __shfl_xor(s, 2, 32);
    s += __shfl_xor(s, 4, 32);
    s += __shfl_xor(s, 8, 32);
    ssa[r] = s;
  }

  const int b = bh >> 4, h = bh & 15;
#pragma unroll
  for (int r = 0; r < 8; ++r) {
    float scale = sqrtf(2048.f) * rsqrtf(ssa[r]);
    size_t orow =
        ((size_t)b * 2048 + qbase + r + ((lane & 16) >> 1)) * 1024 + h * 64;
#pragma unroll
    for (int nt = 0; nt < 4; ++nt)
      ob[orow + nt * 16 + (lane & 15)] = (__bf16)(oacc[nt][r] * scale);
  }
}

// ---------------------------------------------------------------------------
// Elementwise / prep helpers
// ---------------------------------------------------------------------------
__global__ __launch_bounds__(256) void cvt_bf16_kernel(
    const float* __restrict__ in, __bf16* __restrict__ out, size_t n) {
  for (size_t i = (size_t)blockIdx.x * 256 + threadIdx.x; i < n;
       i += (size_t)gridDim.x * 256)
    out[i] = (__bf16)in[i];
}

// (P x H) f32 -> (H x P) bf16 tiled transpose (32x32 tiles via LDS).
__global__ __launch_bounds__(256) void transpose_bf16_kernel(
    const float* __restrict__ in, __bf16* __restrict__ out, int P, int H) {
  __shared__ float t[32][33];
  const int bx = blockIdx.x * 32;        // H offset
  const int by = blockIdx.y * 32;        // P offset
  const int lx = threadIdx.x & 31;
  const int ly = threadIdx.x >> 5;       // 0..7
#pragma unroll
  for (int i = 0; i < 32; i += 8)
    t[ly + i][lx] = in[(size_t)(by + ly + i) * H + bx + lx];
  __syncthreads();
#pragma unroll
  for (int i = 0; i < 32; i += 8)
    out[(size_t)(bx + ly + i) * P + by + lx] = (__bf16)t[lx][ly + i];
}

__global__ __launch_bounds__(256) void add_f32_kernel(
    const float* __restrict__ a, const float* __restrict__ b,
    float* __restrict__ out, size_t n) {
  for (size_t i = (size_t)blockIdx.x * 256 + threadIdx.x; i < n;
       i += (size_t)gridDim.x * 256)
    out[i] = a[i] + b[i];
}

// ---------------------------------------------------------------------------
// Host orchestration
// ---------------------------------------------------------------------------
static const int Bc = 2, Sc = 2048, Hc = 1024, NHc = 16;
static const size_t Mrows = (size_t)Bc * Sc;        // 4096
static const size_t NTOK  = Mrows * Hc;             // 4,194,304

extern "C" void kernel_launch(void* const* d_in, const int* in_sizes, int n_in,
                              void* d_out, int out_size, void* d_ws,
                              size_t ws_size, hipStream_t stream) {
  (void)n_in; (void)out_size; (void)ws_size;
  const float* x    = (const float*)d_in[0];
  const float* ln1w = (const float*)d_in[1];
  const float* ln1b = (const float*)d_in[2];
  const float* ln2w = (const float*)d_in[3];
  const float* ln2b = (const float*)d_in[4];
  const float* wsrc[10];
  size_t wn[10];
  for (int i = 0; i < 10; ++i) {
    wsrc[i] = (const float*)d_in[5 + i];
    wn[i]   = (size_t)in_sizes[5 + i];
  }

  // Workspace arena
  char* p = (char*)d_ws;
  auto alloc = [&](size_t bytes) -> char* {
    char* r = p;
    p += (bytes + 255) & ~(size_t)255;
    return r;
  };
  // wb[even] = key weights (P x H, row-major bf16)
  // wb[odd]  = val weights TRANSPOSED (H x P, row-major bf16)
  __bf16* wb[10];
  for (int i = 0; i < 10; ++i) wb[i] = (__bf16*)alloc(wn[i] * 2);
  __bf16* xnb = (__bf16*)alloc(NTOK * 2);                  // LN output
  float*  aw  = (float*) alloc(Mrows * 4096 * 4);          // slot scores f32
  __bf16* awb = (__bf16*)alloc(Mrows * 4096 * 2);          // normalized bf16
  float*  qf  = (float*) alloc(NTOK * 4);
  float*  kf  = (float*) alloc(NTOK * 4);
  float*  vf  = (float*) alloc(NTOK * 4);
  __bf16* qhB = (__bf16*)alloc(NTOK * 2);
  __bf16* khB = (__bf16*)alloc(NTOK * 2);
  __bf16* vhB = (__bf16*)alloc(NTOK * 2);
  __bf16* obB = (__bf16*)alloc(NTOK * 2);
  float*  x1  = (float*) alloc(NTOK * 4);
  float*  tmp = (float*) alloc(NTOK * 4);

  // 1. Weight prep: keys -> bf16 as-is; vals -> transposed bf16 so the
  //    second pattention GEMM also takes the NT fast path (16B LDS staging,
  //    no transposed scatter in the hot loop).
  for (int i = 0; i < 10; i += 2)
    cvt_bf16_kernel<<<1024, 256, 0, stream>>>(wsrc[i], wb[i], wn[i]);
  for (int i = 1; i < 10; i += 2) {
    int P = (int)(wn[i] / Hc);          // 1024 or 4096 slots
    transpose_bf16_kernel<<<dim3(Hc / 32, P / 32), 256, 0, stream>>>(
        wsrc[i], wb[i], P, Hc);
  }

  // pattention: out = gelu_l2norm(in . key^T * sqrt(H)) . val
  auto pattn = [&](const __bf16* inp, const __bf16* key, const __bf16* valT,
                   float* outF, int P) {
    gemm_bf16_nt_kernel<<<dim3(P / 128, (int)(Mrows / 64)), 256, 0, stream>>>(
        inp, key, aw, P, Hc, 32.0f /* sqrt(1024) */);
    gelu_l2norm_kernel<<<(int)Mrows, 256, 0, stream>>>(aw, awb, P);
    gemm_bf16_nt_kernel<<<dim3(Hc / 128, (int)(Mrows / 64)), 256, 0, stream>>>(
        awb, valT, outF, Hc, P, 1.0f);
  };

  // 2. LN1 + QKV pattentions
  layernorm_bf16_kernel<<<(int)Mrows, 256, 0, stream>>>(x, ln1w, ln1b, xnb);
  pattn(xnb, wb[0], wb[1], qf, 1024);
  pattn(xnb, wb[2], wb[3], kf, 1024);
  pattn(xnb, wb[4], wb[5], vf, 1024);

  // 3. RoPE + head-major repack (bf16)
  rope_pack_kernel<<<(int)(NTOK / 256), 256, 0, stream>>>(qf, kf, vf, qhB, khB,
                                                          vhB);

  // 4. Causal attention (flash-style, single pass)
  attention_kernel<<<dim3(Sc / 64, Bc * NHc), 128, 0, stream>>>(qhB, khB, vhB,
                                                                obB);

  // 5. proj pattention + residual
  pattn(obB, wb[6], wb[7], tmp, 1024);
  add_f32_kernel<<<2048, 256, 0, stream>>>(x, tmp, x1, NTOK);

  // 6. LN2 + FFN pattention + residual -> d_out (f32)
  layernorm_bf16_kernel<<<(int)Mrows, 256, 0, stream>>>(x1, ln2w, ln2b, xnb);
  pattn(xnb, wb[8], wb[9], tmp, 4096);
  add_f32_kernel<<<2048, 256, 0, stream>>>(x1, tmp, (float*)d_out, NTOK);
}
